// CausalSelfAttention_66864050864292
// MI455X (gfx1250) — matrix operand order repaired and checked by reading, not compiled, not run on hardware
//
#include <hip/hip_runtime.h>

// CausalSelfAttention forward for MI455X (gfx1250, wave32, WMMA, async-LDS).
// Pipeline: f32->f16 convert (weights pre-transposed) -> WMMA GEMM (qkv, V written
// transposed) -> flash attention (WMMA) -> WMMA GEMM (proj).
// All streamed tiles are byte-identical copies done with GLOBAL_LOAD_ASYNC_TO_LDS_B128
// (ASYNCcnt) into double-buffered LDS, overlapped with WMMA compute.
// Softmax butterfly reductions use v_permlane16_b32 (VALU) instead of ds_bpermute.

typedef __attribute__((ext_vector_type(16))) _Float16 v16h;
typedef __attribute__((ext_vector_type(8)))  _Float16 v8h;
typedef __attribute__((ext_vector_type(8)))  float    v8f;

#define B_  2
#define T_  4096
#define C_  512
#define H_  8
#define HD_ 64

// Async copy of 16 bytes global -> LDS (per-lane addresses), tracked by ASYNCcnt.
__device__ __forceinline__ void async_copy_b128(const void* gp, const void* lds_flat) {
  unsigned lds = (unsigned)(unsigned long long)lds_flat;  // LDS aperture: low 32 bits = LDS offset
  asm volatile("global_load_async_to_lds_b128 %0, %1, off"
               :: "v"(lds), "v"((unsigned long long)gp)
               : "memory");
}
__device__ __forceinline__ void wait_async0() {
  asm volatile("s_wait_asynccnt 0" ::: "memory");
}

// 16-lane-group butterfly via v_permlane16_b32 (select table = lane ^ mask).
template <unsigned LO, unsigned HI>
__device__ __forceinline__ float permlane16_f(float v) {
  unsigned u = __builtin_bit_cast(unsigned, v);
  unsigned r = __builtin_amdgcn_permlane16(u, u, LO, HI, false, false);
  return __builtin_bit_cast(float, r);
}
__device__ __forceinline__ float redmax16(float v) {
  v = fmaxf(v, permlane16_f<0x67452301u, 0xEFCDAB89u>(v));  // xor 1
  v = fmaxf(v, permlane16_f<0x54761032u, 0xDCFE98BAu>(v));  // xor 2
  v = fmaxf(v, permlane16_f<0x32107654u, 0xBA98FEDCu>(v));  // xor 4
  v = fmaxf(v, permlane16_f<0xFEDCBA98u, 0x76543210u>(v));  // xor 8
  return v;
}
__device__ __forceinline__ float redsum16(float v) {
  v = v + permlane16_f<0x67452301u, 0xEFCDAB89u>(v);
  v = v + permlane16_f<0x54761032u, 0xDCFE98BAu>(v);
  v = v + permlane16_f<0x32107654u, 0xBA98FEDCu>(v);
  v = v + permlane16_f<0xFEDCBA98u, 0x76543210u>(v);
  return v;
}

// ---------------------------------------------------------------- converts
__global__ void f32_to_f16_kernel(const float* __restrict__ src,
                                  _Float16* __restrict__ dst, int n) {
  int i = blockIdx.x * blockDim.x + threadIdx.x;
  if (i < n) dst[i] = (_Float16)src[i];
}
// src [K][N] f32 -> dst [N][K] f16 (weights; one-time, tiny)
__global__ void f32_to_f16_T_kernel(const float* __restrict__ src,
                                    _Float16* __restrict__ dst, int K, int N) {
  int i = blockIdx.x * blockDim.x + threadIdx.x;
  if (i < K * N) {
    int k = i / N, n = i - k * N;
    dst[(size_t)n * K + k] = (_Float16)src[i];
  }
}

// ---------------------------------------------------------------- WMMA GEMM
// C[M,N] = A[M,K](f16) * B(f16, pre-transposed [N][K]) + bias[N](f32)
// Block: 256 threads = 8 waves. Block tile 128(M) x 128(N), K-step 32, double-buffered.
// Wave grid 4(M) x 2(N); each wave owns a 32x64 tile = 2x4 WMMA accumulators.
// Both LDS tiles are byte-identical async copies.
// SPLIT_V: columns [1024,1536) (the V third of qkv) are written transposed to
// VtOut[b][h][hd][t] instead of Cm.
template <typename OutT, bool SPLIT_V>
__global__ __launch_bounds__(256)
void gemm_f16_wmma(const _Float16* __restrict__ A,    // [M][K]
                   const _Float16* __restrict__ BT,   // [N][K]
                   const float* __restrict__ bias, OutT* __restrict__ Cm,
                   _Float16* __restrict__ VtOut,
                   int M, int N, int K) {
  __shared__ __attribute__((aligned(16))) _Float16 As[2][128 * 32];  // [row][k]
  __shared__ __attribute__((aligned(16))) _Float16 Bs[2][128 * 32];  // [col][k]

  const int t      = threadIdx.x;
  const int lane   = t & 31;
  const int w      = t >> 5;
  const int wm     = w >> 1;        // wave M group (0..3)
  const int wn     = w & 1;         // wave N group (0..1)
  const int hiHalf = lane >> 4;     // 0: lanes 0-15, 1: lanes 16-31
  const int ln     = lane & 15;
  const int m0     = blockIdx.x * 128;
  const int n0     = blockIdx.y * 128;
  const int nk     = K >> 5;

  auto loadTiles = [&](int ki, int buf) {
    #pragma unroll
    for (int it = 0; it < 2; ++it) {
      int s = t + it * 256;
      int row = s >> 2, cg = s & 3;
      async_copy_b128(A  + (size_t)(m0 + row) * K + ki * 32 + cg * 8,
                      &As[buf][row * 32 + cg * 8]);
      async_copy_b128(BT + (size_t)(n0 + row) * K + ki * 32 + cg * 8,
                      &Bs[buf][row * 32 + cg * 8]);
    }
  };

  v8f acc[2][4] = {};

  loadTiles(0, 0);
  wait_async0();
  __syncthreads();

  for (int ki = 0; ki < nk; ++ki) {
    const int cur = ki & 1;
    if (ki + 1 < nk) loadTiles(ki + 1, cur ^ 1);

    #pragma unroll
    for (int mi = 0; mi < 2; ++mi) {
      // A fragment 16x32: halves 0-7 -> K off, 8-15 -> K off+16 (per half-wave)
      const _Float16* ap = &As[cur][(wm * 32 + mi * 16 + ln) * 32 + hiHalf * 8];
      v8h alo = *(const v8h*)(ap);
      v8h ahi = *(const v8h*)(ap + 16);
      v16h af = __builtin_shufflevector(alo, ahi, 0,1,2,3,4,5,6,7,8,9,10,11,12,13,14,15);
      #pragma unroll
      for (int ni = 0; ni < 4; ++ni) {
        // B fragment 32x16: halves 0-7 -> K off, 8-15 -> K off+16 (per half-wave)
        const _Float16* bp = &Bs[cur][(wn * 64 + ni * 16 + ln) * 32 + hiHalf * 8];
        v8h blo = *(const v8h*)(bp);
        v8h bhi = *(const v8h*)(bp + 16);
        v16h bf = __builtin_shufflevector(blo, bhi, 0,1,2,3,4,5,6,7,8,9,10,11,12,13,14,15);
        acc[mi][ni] = __builtin_amdgcn_wmma_f32_16x16x32_f16(
            false, af, false, bf, (short)0, acc[mi][ni], false, false);
      }
    }

    wait_async0();
    __syncthreads();
  }

  // Epilogue: D layout -> lane holds col (ln), rows j + 8*hiHalf
  #pragma unroll
  for (int mi = 0; mi < 2; ++mi) {
    #pragma unroll
    for (int ni = 0; ni < 4; ++ni) {
      int colG = n0 + wn * 64 + ni * 16 + ln;
      float bv = bias[colG];
      #pragma unroll
      for (int j = 0; j < 8; ++j) {
        int rowG = m0 + wm * 32 + mi * 16 + j + 8 * hiHalf;
        float val = acc[mi][ni][j] + bv;
        if (SPLIT_V && colG >= 2 * C_) {
          // V element: write transposed as Vt[b][h][hd][t]
          int bb = rowG >> 12, tt = rowG & (T_ - 1);
          int cd = colG - 2 * C_;
          int hh = cd >> 6, dd = cd & (HD_ - 1);
          VtOut[((size_t)(bb * H_ + hh) * HD_ + dd) * T_ + tt] = (_Float16)val;
        } else {
          Cm[(size_t)rowG * N + colG] = (OutT)val;
        }
      }
    }
  }
}

// ---------------------------------------------------------------- flash attention
// One workgroup per (b, h, 64-query tile). 128 threads = 4 waves x 16 queries.
// K/V^T streamed in 32-key blocks, double-buffered, both via async byte-copies.
// Online softmax (permlane reductions); all matmuls via WMMA.
__global__ __launch_bounds__(128)
void flash_attn_kernel(const _Float16* __restrict__ qkv,
                       const _Float16* __restrict__ vtg,   // [b][h][hd][t]
                       _Float16* __restrict__ Y) {
  __shared__ __attribute__((aligned(16))) _Float16 Qs[64 * 64];      // [q][hd] (pre-scaled)
  __shared__ __attribute__((aligned(16))) _Float16 Ks[2][32 * 64];   // [key][hd]
  __shared__ __attribute__((aligned(16))) _Float16 Vt[2][64 * 32];   // [hd][key]
  __shared__ __attribute__((aligned(16))) _Float16 Ps[4][16 * 32];   // per-wave P stage

  const int t      = threadIdx.x;
  const int lane   = t & 31;
  const int w      = t >> 5;
  const int hiHalf = lane >> 4;
  const int ln     = lane & 15;
  const int q0     = blockIdx.x * 64;
  const int b      = blockIdx.y / H_;
  const int h      = blockIdx.y % H_;

  const size_t rs = 3 * C_;  // qkv row stride (halves)
  const _Float16* Qg  = qkv + (size_t)b * T_ * rs + h * HD_;
  const _Float16* Kg  = Qg + C_;
  const _Float16* VtG = vtg + (size_t)(b * H_ + h) * HD_ * T_;

  // Both K (natural) and V^T (transposed in global) are byte-identical async copies.
  auto loadKV = [&](int kb, int buf) {
    #pragma unroll
    for (int it = 0; it < 2; ++it) {
      int s = t + it * 128;
      {
        int row = s >> 3, cg = s & 7;  // 32 keys x 64 hd
        async_copy_b128(Kg + (size_t)(kb * 32 + row) * rs + cg * 8,
                        &Ks[buf][row * 64 + cg * 8]);
      }
      {
        int hd = s >> 2, cg = s & 3;   // 64 hd x 32 keys
        async_copy_b128(VtG + (size_t)hd * T_ + kb * 32 + cg * 8,
                        &Vt[buf][hd * 32 + cg * 8]);
      }
    }
  };

  // Prefetch first K/V block; load Q tile (pre-scaled by 1/sqrt(HD) = 0.125).
  loadKV(0, 0);
  #pragma unroll
  for (int it = 0; it < 4; ++it) {
    int s = t + it * 128;
    int row = s >> 3, cg = s & 7;
    v8h v = *(const v8h*)(Qg + (size_t)(q0 + row) * rs + cg * 8);
    #pragma unroll
    for (int e = 0; e < 8; ++e) v[e] = v[e] * (_Float16)0.125f;
    *(v8h*)(Qs + row * 64 + cg * 8) = v;
  }
  wait_async0();
  __syncthreads();

  // Q A-fragments for the two 32-wide contraction steps over HD
  v16h qf[2];
  #pragma unroll
  for (int ks = 0; ks < 2; ++ks) {
    const _Float16* ap = Qs + (w * 16 + ln) * 64 + ks * 32 + hiHalf * 8;
    v8h lo = *(const v8h*)(ap);
    v8h hi = *(const v8h*)(ap + 16);
    qf[ks] = __builtin_shufflevector(lo, hi, 0,1,2,3,4,5,6,7,8,9,10,11,12,13,14,15);
  }

  v8f o[4] = {};
  float rmax[8], rsum[8];
  #pragma unroll
  for (int j = 0; j < 8; ++j) { rmax[j] = -1e30f; rsum[j] = 0.0f; }

  const int nkb    = q0 / 32 + 2;  // key blocks covering keys 0 .. q0+63
  const int kbDiag = q0 >> 5;      // first block that can touch the diagonal

  for (int kb = 0; kb < nkb; ++kb) {
    const int cur = kb & 1;
    if (kb + 1 < nkb) loadKV(kb + 1, cur ^ 1);

    // S = Q * K^T   (two 16-key sub-tiles, contraction over HD in 2 steps)
    v8f s0 = {}, s1 = {};
    #pragma unroll
    for (int ks = 0; ks < 2; ++ks) {
      const _Float16* bp0 = &Ks[cur][ln * 64 + ks * 32 + hiHalf * 8];
      v8h l0 = *(const v8h*)(bp0);
      v8h h0 = *(const v8h*)(bp0 + 16);
      v16h bf0 = __builtin_shufflevector(l0, h0, 0,1,2,3,4,5,6,7,8,9,10,11,12,13,14,15);
      s0 = __builtin_amdgcn_wmma_f32_16x16x32_f16(false, qf[ks], false, bf0, (short)0, s0, false, false);
      const _Float16* bp1 = &Ks[cur][(16 + ln) * 64 + ks * 32 + hiHalf * 8];
      v8h l1 = *(const v8h*)(bp1);
      v8h h1 = *(const v8h*)(bp1 + 16);
      v16h bf1 = __builtin_shufflevector(l1, h1, 0,1,2,3,4,5,6,7,8,9,10,11,12,13,14,15);
      s1 = __builtin_amdgcn_wmma_f32_16x16x32_f16(false, qf[ks], false, bf1, (short)0, s1, false, false);
    }

    // Causal mask (only blocks touching the diagonal; scalar-uniform condition)
    // + online softmax. Rows live in VGPR index j + 8*hiHalf, cols across 16 lanes.
    float p0[8], p1[8], fac[8];
    const bool needMask = (kb >= kbDiag);
    #pragma unroll
    for (int j = 0; j < 8; ++j) {
      float a = s0[j], c = s1[j];
      if (needMask) {
        int qi  = q0 + w * 16 + j + 8 * hiHalf;
        int k0i = kb * 32 + ln;
        if (k0i > qi)      a = -1e30f;
        if (k0i + 16 > qi) c = -1e30f;
      }
      float mx = redmax16(fmaxf(a, c));
      mx = fmaxf(mx, rmax[j]);
      float e0 = __expf(a - mx);
      float e1 = __expf(c - mx);
      float srow = redsum16(e0 + e1);
      fac[j]  = __expf(rmax[j] - mx);
      rsum[j] = rsum[j] * fac[j] + srow;
      rmax[j] = mx;
      p0[j] = e0; p1[j] = e1;
    }
    #pragma unroll
    for (int n = 0; n < 4; ++n)
      #pragma unroll
      for (int j = 0; j < 8; ++j) o[n][j] *= fac[j];

    // Stage P (D layout -> A layout) through per-wave LDS; same-wave DS ops are in-order
    _Float16* Pw = Ps[w];
    #pragma unroll
    for (int j = 0; j < 8; ++j) {
      int row = j + 8 * hiHalf;
      Pw[row * 32 + ln]      = (_Float16)p0[j];
      Pw[row * 32 + 16 + ln] = (_Float16)p1[j];
    }
    const _Float16* pp = Pw + ln * 32 + hiHalf * 8;
    v8h plo = *(const v8h*)(pp);
    v8h phi = *(const v8h*)(pp + 16);
    v16h pf = __builtin_shufflevector(plo, phi, 0,1,2,3,4,5,6,7,8,9,10,11,12,13,14,15);

    // O += P * V   (4 HD sub-tiles of 16 cols)
    #pragma unroll
    for (int n = 0; n < 4; ++n) {
      const _Float16* vp = &Vt[cur][(n * 16 + ln) * 32 + hiHalf * 8];
      v8h vlo = *(const v8h*)(vp);
      v8h vhi = *(const v8h*)(vp + 16);
      v16h vf = __builtin_shufflevector(vlo, vhi, 0,1,2,3,4,5,6,7,8,9,10,11,12,13,14,15);
      o[n] = __builtin_amdgcn_wmma_f32_16x16x32_f16(false, pf, false, vf, (short)0, o[n], false, false);
    }

    wait_async0();
    __syncthreads();
  }

  // Normalize and store y[b, t, h*64 + d] as f16
  #pragma unroll
  for (int n = 0; n < 4; ++n) {
    #pragma unroll
    for (int j = 0; j < 8; ++j) {
      int rowG = q0 + w * 16 + j + 8 * hiHalf;
      int colG = h * HD_ + n * 16 + ln;
      Y[(size_t)(b * T_ + rowG) * C_ + colG] = (_Float16)(o[n][j] / rsum[j]);
    }
  }
}

// ---------------------------------------------------------------- launch
extern "C" void kernel_launch(void* const* d_in, const int* in_sizes, int n_in,
                              void* d_out, int out_size, void* d_ws, size_t ws_size,
                              hipStream_t stream) {
  (void)in_sizes; (void)n_in; (void)out_size; (void)ws_size;
  const float* x     = (const float*)d_in[0];
  const float* Wqkv  = (const float*)d_in[1];
  const float* bqkv  = (const float*)d_in[2];
  const float* Wproj = (const float*)d_in[3];
  const float* bproj = (const float*)d_in[4];
  float* out = (float*)d_out;

  // Workspace layout (f16), all 256B-aligned:
  char* ws = (char*)d_ws;
  _Float16* xh     = (_Float16*)(ws);                    //  8,388,608 B
  _Float16* WqkvT  = (_Float16*)(ws + 8388608);          //  1,572,864 B  [1536][512]
  _Float16* WprojT = (_Float16*)(ws + 9961472);          //    524,288 B  [512][512]
  _Float16* qkvh   = (_Float16*)(ws + 10485760);         // 25,165,824 B  (V third unused)
  _Float16* yh     = (_Float16*)(ws + 35651584);         //  8,388,608 B
  _Float16* vtg    = (_Float16*)(ws + 44040192);         //  8,388,608 B  [b][h][hd][t]
                                                         //  end 52,428,800

  const int nx  = B_ * T_ * C_;      // 4,194,304
  const int nwq = C_ * 3 * C_;       //   786,432
  const int nwp = C_ * C_;           //   262,144

  f32_to_f16_kernel  <<<(nx  + 255) / 256, 256, 0, stream>>>(x, xh, nx);
  f32_to_f16_T_kernel<<<(nwq + 255) / 256, 256, 0, stream>>>(Wqkv,  WqkvT,  C_, 3 * C_);
  f32_to_f16_T_kernel<<<(nwp + 255) / 256, 256, 0, stream>>>(Wproj, WprojT, C_, C_);

  // qkv = x * Wqkv + bqkv    [8192 x 1536]; V third written transposed to vtg
  gemm_f16_wmma<_Float16, true><<<dim3(8192 / 128, 1536 / 128), 256, 0, stream>>>(
      xh, WqkvT, bqkv, qkvh, vtg, B_ * T_, 3 * C_, C_);

  // flash attention -> yh [8192 x 512]
  flash_attn_kernel<<<dim3(T_ / 64, B_ * H_), 128, 0, stream>>>(qkvh, vtg, yh);

  // out = y * Wproj + bproj  [8192 x 512]
  gemm_f16_wmma<float, false><<<dim3(8192 / 128, 512 / 128), 256, 0, stream>>>(
      yh, WprojT, bproj, out, nullptr, B_ * T_, C_, C_);
}
